// SlotAttention_90812788506749
// MI455X (gfx1250) — matrix operand order repaired and checked by reading, not compiled, not run on hardware
//
#include <hip/hip_runtime.h>
#include <hip/hip_bf16.h>

#define BB 32
#define NN 4096
#define CC 256
#define DD 64
#define SS 7
#define HH 128
#define KSPLIT 16   // grid-level K split for the attention GEMM

typedef __attribute__((ext_vector_type(16))) __bf16 v16bf;
typedef __attribute__((ext_vector_type(8)))  float  v8f;
typedef __attribute__((ext_vector_type(4)))  unsigned int uint4v;

union BF16Frag { uint4v q[2]; v16bf v; };
union BF16Pack8 { uint4v q; unsigned short s[8]; };

__device__ __forceinline__ unsigned short f2bf(float f) {
    unsigned u = __builtin_bit_cast(unsigned, f);
    u += 0x7FFFu + ((u >> 16) & 1u);           // round-to-nearest-even
    return (unsigned short)(u >> 16);
}

// ---------------------------------------------------------------------------
// prep: fp32 Wkv -> bf16, copy slots into mutable workspace
// ---------------------------------------------------------------------------
__global__ void prep_kernel(const float* __restrict__ Wkv, const float* __restrict__ slots_in,
                            unsigned short* __restrict__ Wkv_bf, float* __restrict__ slots_ws) {
    int i = blockIdx.x * blockDim.x + threadIdx.x;
    if (i < 2 * DD * CC) Wkv_bf[i] = f2bf(Wkv[i]);
    if (i < BB * SS * DD) slots_ws[i] = slots_in[i];
}

// ---------------------------------------------------------------------------
// Fused LayerNorm(x) + kv GEMM (bf16 WMMA). Block = 64 rows x 128 cols tile.
// k stored (B,N,D) via LDS-transposed coalesced b128 stores;
// v stored transposed (B,D,N) via packed b128 stores.
// ---------------------------------------------------------------------------
__global__ __launch_bounds__(256) void lnkv_kernel(const float* __restrict__ x,
                                                   const float* __restrict__ g_in,
                                                   const float* __restrict__ b_in,
                                                   const unsigned short* __restrict__ Wkv_bf,
                                                   unsigned short* __restrict__ k_bf,
                                                   unsigned short* __restrict__ vT_bf) {
    __shared__ unsigned short xs[64][CC + 8];   // bf16 LN(x) tile, padded
    __shared__ unsigned short ks[64][DD + 8];   // k tile staging for coalesced store
    const int b    = blockIdx.y;
    const int n0   = blockIdx.x * 64;
    const int lane = threadIdx.x & 31;
    const int wv   = threadIdx.x >> 5;

    // --- Phase 1: LayerNorm of 8 rows per wave (row length 256) ---
    for (int i = 0; i < 8; ++i) {
        const int row = wv * 8 + i;
        const float* xr = x + ((size_t)(b * NN + n0 + row)) * CC;
        float v[8], s = 0.f, sq = 0.f;
        #pragma unroll
        for (int j = 0; j < 8; ++j) { v[j] = xr[lane + j * 32]; s += v[j]; sq += v[j] * v[j]; }
        #pragma unroll
        for (int off = 16; off > 0; off >>= 1) {
            s  += __shfl_xor(s,  off, 32);
            sq += __shfl_xor(sq, off, 32);
        }
        const float mean = s * (1.0f / CC);
        const float var  = sq * (1.0f / CC) - mean * mean;
        const float rstd = rsqrtf(var + 1e-5f);
        #pragma unroll
        for (int j = 0; j < 8; ++j) {
            const int c = lane + j * 32;
            xs[row][c] = f2bf((v[j] - mean) * rstd * g_in[c] + b_in[c]);
        }
    }
    __syncthreads();

    // --- Phase 2: WMMA 64x128 = xs(64x256) @ Wkv^T(256x128) ---
    const int mt = wv >> 1;                 // M tile 0..3
    const int nh = wv & 1;                  // N half: 0 -> k, 1 -> v
    v8f acc[4];
    #pragma unroll
    for (int t = 0; t < 4; ++t) acc[t] = (v8f){0.f,0.f,0.f,0.f,0.f,0.f,0.f,0.f};

    const int m     = mt * 16 + (lane & 15);
    const int ncol  = lane & 15;
    const int ahalf = (lane & 16) ? 16 : 0; // A: 16 contiguous halfs
    const int bhalf = (lane & 16) ? 16 : 0; // B: 16 contiguous halfs

    for (int ch = 0; ch < 8; ++ch) {        // K = 256 in chunks of 32
        BF16Frag A;
        const int ka = ch * 32 + ahalf;
        A.q[0] = *(const uint4v*)&xs[m][ka];
        A.q[1] = *(const uint4v*)&xs[m][ka + 8];
        #pragma unroll
        for (int t = 0; t < 4; ++t) {
            const int e  = nh * 64 + t * 16 + ncol;
            const unsigned short* wp = Wkv_bf + (size_t)e * CC + ch * 32 + bhalf;
            BF16Frag Bf;
            Bf.q[0] = *(const uint4v*)wp;
            Bf.q[1] = *(const uint4v*)(wp + 8);
            acc[t] = __builtin_amdgcn_wmma_f32_16x16x32_bf16(
                false, A.v, false, Bf.v, (short)0, acc[t], false, false);
        }
    }

    // --- Store ---
    const int mrow_base = mt * 16 + ((lane & 16) ? 8 : 0);
    if (nh == 0) {
        // k tiles -> LDS staging (transpose), coalesced store after barrier
        #pragma unroll
        for (int t = 0; t < 4; ++t) {
            const int e = t * 16 + ncol;
            #pragma unroll
            for (int r = 0; r < 8; ++r) ks[mrow_base + r][e] = f2bf(acc[t][r]);
        }
    } else {
        // v tiles: 8 consecutive positions per lane -> one b128 store each
        #pragma unroll
        for (int t = 0; t < 4; ++t) {
            const int d = t * 16 + ncol;
            BF16Pack8 P;
            #pragma unroll
            for (int r = 0; r < 8; ++r) P.s[r] = f2bf(acc[t][r]);
            *(uint4v*)&vT_bf[((size_t)b * DD + d) * NN + n0 + mrow_base] = P.q;
        }
    }
    __syncthreads();
    {   // coalesced k store: 64 rows x 4 x 8 ushorts (at part*16) = 256 x 16B
        const int row  = threadIdx.x >> 2;
        const int part = threadIdx.x & 3;
        *(uint4v*)&k_bf[((size_t)(b * NN) + n0 + row) * DD + part * 16] =
            *(const uint4v*)&ks[row][part * 16];
    }
}

// ---------------------------------------------------------------------------
// q = LN(slots) @ Wq^T -> bf16 padded to 16 rows; zeroes sums and attn accum.
// ---------------------------------------------------------------------------
__global__ __launch_bounds__(256) void q_kernel(const float* __restrict__ slots,
                                                const float* __restrict__ g_sl,
                                                const float* __restrict__ b_sl,
                                                const float* __restrict__ Wq,
                                                unsigned short* __restrict__ q_bf,
                                                float* __restrict__ sums,
                                                float* __restrict__ attn) {
    __shared__ float lnS[SS][DD];
    const int b = blockIdx.x;
    const int lane = threadIdx.x & 31;
    const int wv   = threadIdx.x >> 5;
    if (wv < SS) {
        const float* sp = slots + (b * SS + wv) * DD;
        const float v0 = sp[lane], v1 = sp[lane + 32];
        float s = v0 + v1, sq = v0 * v0 + v1 * v1;
        #pragma unroll
        for (int off = 16; off > 0; off >>= 1) {
            s  += __shfl_xor(s,  off, 32);
            sq += __shfl_xor(sq, off, 32);
        }
        const float mean = s * (1.0f / DD);
        const float var  = sq * (1.0f / DD) - mean * mean;
        const float rstd = rsqrtf(var + 1e-5f);
        lnS[wv][lane]      = (v0 - mean) * rstd * g_sl[lane]      + b_sl[lane];
        lnS[wv][lane + 32] = (v1 - mean) * rstd * g_sl[lane + 32] + b_sl[lane + 32];
    }
    __syncthreads();
    for (int idx = threadIdx.x; idx < SS * DD; idx += 256) {
        const int s = idx >> 6, e = idx & 63;
        float a = 0.f;
        #pragma unroll 8
        for (int d = 0; d < DD; ++d) a += lnS[s][d] * Wq[e * DD + d];
        q_bf[b * 16 * DD + idx] = f2bf(a);
    }
    for (int idx = threadIdx.x + SS * DD; idx < 16 * DD; idx += 256)
        q_bf[b * 16 * DD + idx] = 0;                     // pad rows 7..15
    for (int idx = threadIdx.x; idx < SS * DD; idx += 256)
        attn[b * SS * DD + idx] = 0.f;                   // attn accumulator
    if (threadIdx.x < SS) sums[b * SS + threadIdx.x] = 0.f;
}

// ---------------------------------------------------------------------------
// logits = q @ k^T (WMMA), softmax over S per column, store w (bf16, padded
// to 16 rows), atomic partial sums over N per slot.
// ---------------------------------------------------------------------------
__global__ __launch_bounds__(128) void logits_kernel(const unsigned short* __restrict__ q_bf,
                                                     const unsigned short* __restrict__ k_bf,
                                                     unsigned short* __restrict__ w_bf,
                                                     float* __restrict__ sums) {
    const int b    = blockIdx.y;
    const int lane = threadIdx.x & 31;
    const int wv   = threadIdx.x >> 5;
    const int p0   = blockIdx.x * 64 + wv * 16;     // 16 positions per wave

    v8f c = (v8f){0.f,0.f,0.f,0.f,0.f,0.f,0.f,0.f};
    const int m     = lane & 15;
    const int ahalf = (lane & 16) ? 16 : 0;
    const int bhalf = (lane & 16) ? 16 : 0;
    const int p     = p0 + (lane & 15);

    #pragma unroll
    for (int ch = 0; ch < 2; ++ch) {                // K = 64
        BF16Frag A, Bf;
        const unsigned short* qp = q_bf + b * 16 * DD + m * DD + ch * 32 + ahalf;
        A.q[0] = *(const uint4v*)qp;
        A.q[1] = *(const uint4v*)(qp + 8);
        const unsigned short* kp = k_bf + ((size_t)(b * NN) + p) * DD + ch * 32 + bhalf;
        Bf.q[0] = *(const uint4v*)kp;
        Bf.q[1] = *(const uint4v*)(kp + 8);
        c = __builtin_amdgcn_wmma_f32_16x16x32_bf16(false, A.v, false, Bf.v, (short)0, c,
                                                    false, false);
    }

    // lane n (0..15) holds logits for s=0..6 of column p in c[0..6]
    const float scale = 0.125f;                     // 1/sqrt(64)
    float l[SS];
    float mx = -1e30f;
    #pragma unroll
    for (int r = 0; r < SS; ++r) { l[r] = c[r] * scale; mx = fmaxf(mx, l[r]); }
    float ssum = 0.f;
    #pragma unroll
    for (int r = 0; r < SS; ++r) { l[r] = __expf(l[r] - mx); ssum += l[r]; }
    const float inv = 1.0f / ssum;
    #pragma unroll
    for (int r = 0; r < SS; ++r) l[r] *= inv;

    if (lane < 16) {
        unsigned short* wp = w_bf + (size_t)b * 16 * NN + p;
        #pragma unroll
        for (int s = 0; s < SS; ++s) wp[(size_t)s * NN] = f2bf(l[s]);
        #pragma unroll
        for (int s = SS; s < 16; ++s) wp[(size_t)s * NN] = 0;
    }
    #pragma unroll
    for (int s = 0; s < SS; ++s) {                  // Σ over this tile's 16 columns
        float v = l[s];
        v += __shfl_xor(v, 8, 16);
        v += __shfl_xor(v, 4, 16);
        v += __shfl_xor(v, 2, 16);
        v += __shfl_xor(v, 1, 16);
        if (lane == 0) atomicAdd(&sums[b * SS + s], v);
    }
}

// ---------------------------------------------------------------------------
// attn += (w / (sum+eps)) @ v  — WMMA, K split KSPLIT ways across the grid for
// bandwidth-saturating concurrency; partials combined with fp32 atomics.
// ---------------------------------------------------------------------------
__global__ __launch_bounds__(128) void attn_kernel(const unsigned short* __restrict__ w_bf,
                                                   const unsigned short* __restrict__ vT_bf,
                                                   const float* __restrict__ sums,
                                                   float* __restrict__ attn) {
    const int b    = blockIdx.y;
    const int kseg = blockIdx.x;                    // 0..KSPLIT-1
    const int lane = threadIdx.x & 31;
    const int dt   = threadIdx.x >> 5;              // 16-wide D tile per wave

    v8f c = (v8f){0.f,0.f,0.f,0.f,0.f,0.f,0.f,0.f};
    const int m     = lane & 15;
    const int ahalf = (lane & 16) ? 16 : 0;
    const int bhalf = (lane & 16) ? 16 : 0;
    const int d     = dt * 16 + (lane & 15);
    const int nbase = kseg * (NN / KSPLIT);
    const unsigned short* arow = w_bf + (size_t)b * 16 * NN + (size_t)m * NN + nbase;
    const unsigned short* brow = vT_bf + ((size_t)b * DD + d) * NN + nbase;

    #pragma unroll
    for (int ch = 0; ch < NN / KSPLIT / 32; ++ch) { // 8 chunks of K=32
        BF16Frag A, Bf;
        const unsigned short* ap = arow + ch * 32 + ahalf;
        A.q[0] = *(const uint4v*)ap;
        A.q[1] = *(const uint4v*)(ap + 8);
        const unsigned short* bp = brow + ch * 32 + bhalf;
        Bf.q[0] = *(const uint4v*)bp;
        Bf.q[1] = *(const uint4v*)(bp + 8);
        c = __builtin_amdgcn_wmma_f32_16x16x32_bf16(false, A.v, false, Bf.v, (short)0, c,
                                                    false, false);
    }
    if (lane < 16) {                                // rows 0..6 live in lanes 0-15
        #pragma unroll
        for (int r = 0; r < SS; ++r) {
            const float rec = 1.0f / (sums[b * SS + r] + 1e-12f);
            atomicAdd(&attn[(b * SS + r) * DD + dt * 16 + lane], c[r] * rec);
        }
    }
}

// ---------------------------------------------------------------------------
// GRU gates + residual MLP, one block per (b,s); tiny fp32 work.
// ---------------------------------------------------------------------------
__global__ __launch_bounds__(64) void gru_mlp_kernel(const float* __restrict__ attn,
                                                     float* __restrict__ slots,
                                                     const float* __restrict__ Wih,
                                                     const float* __restrict__ Whh,
                                                     const float* __restrict__ bih,
                                                     const float* __restrict__ bhh,
                                                     const float* __restrict__ g_mlp,
                                                     const float* __restrict__ b_mlp,
                                                     const float* __restrict__ W1,
                                                     const float* __restrict__ b1,
                                                     const float* __restrict__ W2,
                                                     const float* __restrict__ b2) {
    const int s = blockIdx.x, b = blockIdx.y;
    const int d = threadIdx.x;
    __shared__ float at[DD], sl[DD], gi[3 * DD], gh[3 * DD], lnv[DD], hb[HH];
    __shared__ float red[DD], red2[DD];

    const float* ap = attn + (b * SS + s) * DD;
    float* sp = slots + (b * SS + s) * DD;
    at[d] = ap[d];
    sl[d] = sp[d];
    __syncthreads();

    #pragma unroll
    for (int j = 0; j < 3; ++j) {
        const int g = j * DD + d;
        float a1 = bih[g], a2 = bhh[g];
        #pragma unroll 8
        for (int d2 = 0; d2 < DD; ++d2) {
            a1 += at[d2] * Wih[g * DD + d2];
            a2 += sl[d2] * Whh[g * DD + d2];
        }
        gi[g] = a1; gh[g] = a2;
    }
    __syncthreads();

    const float r  = 1.0f / (1.0f + __expf(-(gi[d] + gh[d])));
    const float z  = 1.0f / (1.0f + __expf(-(gi[DD + d] + gh[DD + d])));
    const float nn = tanhf(gi[2 * DD + d] + r * gh[2 * DD + d]);
    const float snew = (1.0f - z) * nn + z * sl[d];
    red[d] = snew; red2[d] = snew * snew;
    __syncthreads();
    for (int o = 32; o > 0; o >>= 1) {
        if (d < o) { red[d] += red[d + o]; red2[d] += red2[d + o]; }
        __syncthreads();
    }
    const float mean = red[0] * (1.0f / DD);
    const float var  = red2[0] * (1.0f / DD) - mean * mean;
    const float rstd = rsqrtf(var + 1e-5f);
    lnv[d] = (snew - mean) * rstd * g_mlp[d] + b_mlp[d];
    __syncthreads();

    #pragma unroll
    for (int j = 0; j < 2; ++j) {
        const int h = j * DD + d;
        float a = b1[h];
        #pragma unroll 8
        for (int d2 = 0; d2 < DD; ++d2) a += lnv[d2] * W1[h * DD + d2];
        hb[h] = fmaxf(a, 0.f);
    }
    __syncthreads();

    float out = snew + b2[d];
    #pragma unroll 8
    for (int j = 0; j < HH; ++j) out += hb[j] * W2[d * HH + j];
    sp[d] = out;
}

// ---------------------------------------------------------------------------
extern "C" void kernel_launch(void* const* d_in, const int* in_sizes, int n_in,
                              void* d_out, int out_size, void* d_ws, size_t ws_size,
                              hipStream_t stream) {
    const float* x     = (const float*)d_in[0];
    const float* slots = (const float*)d_in[1];
    const float* g_in  = (const float*)d_in[2];
    const float* b_in  = (const float*)d_in[3];
    const float* g_sl  = (const float*)d_in[4];
    const float* b_sl  = (const float*)d_in[5];
    const float* g_mlp = (const float*)d_in[6];
    const float* b_mlp = (const float*)d_in[7];
    const float* Wq    = (const float*)d_in[8];
    const float* Wkv   = (const float*)d_in[9];
    const float* Wih   = (const float*)d_in[10];
    const float* Whh   = (const float*)d_in[11];
    const float* bih   = (const float*)d_in[12];
    const float* bhh   = (const float*)d_in[13];
    const float* W1    = (const float*)d_in[14];
    const float* b1    = (const float*)d_in[15];
    const float* W2    = (const float*)d_in[16];
    const float* b2    = (const float*)d_in[17];
    // d_in[18] = num_iters (device scalar; reference default is 3)

    char* ws = (char*)d_ws;
    size_t off = 0;
    auto carve = [&](size_t bytes) { size_t o = off; off += (bytes + 255) & ~(size_t)255; return o; };
    unsigned short* k_bf    = (unsigned short*)(ws + carve((size_t)BB * NN * DD * 2));
    unsigned short* vT_bf   = (unsigned short*)(ws + carve((size_t)BB * DD * NN * 2));
    unsigned short* Wkv_bf  = (unsigned short*)(ws + carve((size_t)2 * DD * CC * 2));
    unsigned short* q_bf    = (unsigned short*)(ws + carve((size_t)BB * 16 * DD * 2));
    unsigned short* w_bf    = (unsigned short*)(ws + carve((size_t)BB * 16 * NN * 2));
    float* sums             = (float*)(ws + carve((size_t)BB * SS * 4));
    float* slots_ws         = (float*)(ws + carve((size_t)BB * SS * DD * 4));
    float* attn             = (float*)(ws + carve((size_t)BB * SS * DD * 4));

    prep_kernel<<<(2 * DD * CC + 255) / 256, 256, 0, stream>>>(Wkv, slots, Wkv_bf, slots_ws);
    lnkv_kernel<<<dim3(NN / 64, BB), 256, 0, stream>>>(x, g_in, b_in, Wkv_bf, k_bf, vT_bf);

    const int num_iters = 3;
    for (int it = 0; it < num_iters; ++it) {
        q_kernel<<<BB, 256, 0, stream>>>(slots_ws, g_sl, b_sl, Wq, q_bf, sums, attn);
        logits_kernel<<<dim3(NN / 64, BB), 128, 0, stream>>>(q_bf, k_bf, w_bf, sums);
        attn_kernel<<<dim3(KSPLIT, BB), 128, 0, stream>>>(w_bf, vT_bf, sums, attn);
        gru_mlp_kernel<<<dim3(SS, BB), 64, 0, stream>>>(attn, slots_ws, Wih, Whh, bih, bhh,
                                                        g_mlp, b_mlp, W1, b1, W2, b2);
    }
    hipMemcpyAsync(d_out, slots_ws, (size_t)BB * SS * DD * sizeof(float),
                   hipMemcpyDeviceToDevice, stream);
}
